// MixtureOfDepthsBridge_3530463117438
// MI455X (gfx1250) — matrix operand order repaired and checked by reading, not compiled, not run on hardware
//
#include <hip/hip_runtime.h>

// ---------------------------------------------------------------------------
// MixtureOfDepths perceiver bridge for MI455X (gfx1250, wave32, WMMA).
// All matmuls run through one bf16 WMMA GEMM (v_wmma_f32_16x16x32_bf16,
// fp32 accumulate), 128x128 block tile, 8 waves (32x64 per wave -> 8 WMMA
// per K-step), double-buffered LDS, async global->LDS (ASYNCcnt) for the
// bf16 weight operand. fp32 A-operands are converted to bf16 while staging.
// Needs ~420MB of d_ws: keys/K/V fp32 (3 x 128MB) + staging + latents.
// ---------------------------------------------------------------------------

typedef __bf16 bf16_t;
typedef __attribute__((ext_vector_type(16))) __bf16 v16bf;
typedef __attribute__((ext_vector_type(8)))  float  v8f;

#if defined(__gfx1250__) && __has_builtin(__builtin_amdgcn_global_load_async_to_lds_b128) && __has_builtin(__builtin_amdgcn_s_wait_asynccnt)
#define USE_ASYNC 1
typedef int v4i_vs __attribute__((vector_size(16)));                 // matches builtin's pointee type
typedef __attribute__((address_space(1))) v4i_vs gas_v4i;            // global (AS1)
typedef __attribute__((address_space(3))) v4i_vs las_v4i;            // LDS (AS3)
#else
#define USE_ASYNC 0
#endif

#define TM 128
#define TN 128
#define TK 32
#define ASTR 40   // padded LDS row stride (bf16): 80B = 20 banks -> conflict-free b128 frag reads
#define BSTR 40

// C[M,N] = A(fp32, MxK row-major) @ W(bf16, NxK row-major)^T + bias (+ residual if RSD)
// Requires M%128==0, N%128==0, K%32==0.
template <bool RSD>
__global__ __launch_bounds__(256)
void wmma_gemm(const float* __restrict__ A, const bf16_t* __restrict__ W,
               const float* __restrict__ bias, const float* __restrict__ Rsd,
               float* __restrict__ C, int M, int N, int K)
{
    __shared__ __align__(16) bf16_t As[2][TM * ASTR];
    __shared__ __align__(16) bf16_t Bs[2][TN * BSTR];

    const int t     = threadIdx.x;
    const int tileM = blockIdx.y * TM;
    const int tileN = blockIdx.x * TN;
    const int wave  = t >> 5;
    const int lane  = t & 31;
    const int wm    = (wave & 3) << 5;   // 0,32,64,96  (M)
    const int wn    = (wave >> 2) << 6;  // 0,64        (N)
    const int fr    = lane & 15;
    const int hi    = lane >> 4;         // 0/1

    v8f acc[2][4] = {};

    // loader mapping: 256 threads, 16 elements each, for a 128x32 tile
    const int lrow  = t >> 1;            // 0..127
    const int lhalf = (t & 1) * 16;      // 0 or 16

    const int nIter = K / TK;

    float4 af[4];      // prefetched A (fp32)
#if !USE_ASYNC
    uint4  bfp[2];     // prefetched B (bf16) register fallback
#endif

    auto loadA = [&](int it) {
        const float4* s4 = (const float4*)(A + (size_t)(tileM + lrow) * K + it * TK + lhalf);
        af[0] = s4[0]; af[1] = s4[1]; af[2] = s4[2]; af[3] = s4[3];
    };
    auto storeA = [&](int buf) {
        union { bf16_t h[16]; uint4 q[2]; } tmp;
        const float* fs = (const float*)af;
        #pragma unroll
        for (int j = 0; j < 16; ++j) tmp.h[j] = (bf16_t)fs[j];
        uint4* dst = (uint4*)&As[buf][lrow * ASTR + lhalf];
        dst[0] = tmp.q[0];
        dst[1] = tmp.q[1];
    };
    auto loadB = [&](int it, int buf) {
        const bf16_t* g = W + (size_t)(tileN + lrow) * K + it * TK + lhalf;
#if USE_ASYNC
        bf16_t* l = &Bs[buf][lrow * BSTR + lhalf];
        __builtin_amdgcn_global_load_async_to_lds_b128((gas_v4i*)g, (las_v4i*)l, 0, 0);
        __builtin_amdgcn_global_load_async_to_lds_b128((gas_v4i*)(g + 8), (las_v4i*)(l + 8), 0, 0);
#else
        const uint4* s = (const uint4*)g;
        bfp[0] = s[0]; bfp[1] = s[1];
#endif
    };
#if !USE_ASYNC
    auto storeB = [&](int buf) {
        uint4* dst = (uint4*)&Bs[buf][lrow * BSTR + lhalf];
        dst[0] = bfp[0];
        dst[1] = bfp[1];
    };
#endif

    union Frag { v16bf v; uint4 q[2]; };

    // prologue: stage 0
    loadA(0);
    loadB(0, 0);
    storeA(0);
#if !USE_ASYNC
    storeB(0);
#else
    __builtin_amdgcn_s_wait_asynccnt(0);
#endif
    __syncthreads();

    for (int it = 0; it < nIter; ++it) {
        const int cur = it & 1;
        const int nxt = cur ^ 1;
        const bool pf = (it + 1) < nIter;
        if (pf) {
            loadA(it + 1);
            loadB(it + 1, nxt);   // async path writes LDS[nxt] directly
        }

        Frag a[2], b[4];
        #pragma unroll
        for (int mi = 0; mi < 2; ++mi) {
            const bf16_t* p = &As[cur][(wm + mi * 16 + fr) * ASTR];
            // A 16x32 bf16 frag: lanes 0-15 k=0..7 & 16..23, lanes 16-31 +8
            a[mi].q[0] = *(const uint4*)(p + hi * 8);
            a[mi].q[1] = *(const uint4*)(p + 16 + hi * 8);
        }
        #pragma unroll
        for (int ni = 0; ni < 4; ++ni) {
            const bf16_t* p = &Bs[cur][(wn + ni * 16 + fr) * BSTR];
            // B 32x16 bf16 frag: lanes 0-15 k=0..7 & 16..23, lanes 16-31 +8
            b[ni].q[0] = *(const uint4*)(p + hi * 8);
            b[ni].q[1] = *(const uint4*)(p + 16 + hi * 8);
        }
        #pragma unroll
        for (int mi = 0; mi < 2; ++mi)
            #pragma unroll
            for (int ni = 0; ni < 4; ++ni)
                acc[mi][ni] = __builtin_amdgcn_wmma_f32_16x16x32_bf16(
                    false, a[mi].v, false, b[ni].v, (short)0, acc[mi][ni], false, false);

        if (pf) {
            storeA(nxt);
#if !USE_ASYNC
            storeB(nxt);
#else
            __builtin_amdgcn_s_wait_asynccnt(0);
#endif
        }
        __syncthreads();
    }

    // epilogue: C/D layout -> lane fr = N col, vgpr r = M row (+8 for hi lanes)
    #pragma unroll
    for (int mi = 0; mi < 2; ++mi) {
        #pragma unroll
        for (int ni = 0; ni < 4; ++ni) {
            const int gm0 = tileM + wm + mi * 16 + hi * 8;
            const int gn  = tileN + wn + ni * 16 + fr;
            const float bv = bias[gn];
            float* cp = C + (size_t)gm0 * N + gn;
            const float* rp = RSD ? (Rsd + (size_t)gm0 * N + gn) : nullptr;
            #pragma unroll
            for (int r = 0; r < 8; ++r) {
                float v = acc[mi][ni][r] + bv;
                if (RSD) v += rp[(size_t)r * N];
                cp[(size_t)r * N] = v;
            }
        }
    }
}

__global__ void cvt_bf16(const float* __restrict__ s, bf16_t* __restrict__ d, int n)
{
    int i = blockIdx.x * 256 + threadIdx.x;
    if (i < n) d[i] = (bf16_t)s[i];
}

__global__ void init_state(const float* __restrict__ latents, float* __restrict__ x,
                           float* __restrict__ acc, int* __restrict__ activeb,
                           float* __restrict__ capacc)
{
    int i = blockIdx.x * 256 + threadIdx.x;
    if (i < 131072) { x[i] = latents[i & 8191]; acc[i] = 0.f; }
    if (i < 128) activeb[i] = 1;
    if (i == 0) capacc[0] = 0.f;
}

__global__ __launch_bounds__(256)
void router_k(const float* __restrict__ x, const float* __restrict__ w,
              const float* __restrict__ b0, float* __restrict__ scores)
{
    int row = blockIdx.x, t = threadIdx.x;
    __shared__ float red[256];
    const float* xr = x + (size_t)row * 1024;
    float s = 0.f;
    for (int d = t; d < 1024; d += 256) s += xr[d] * w[d];
    red[t] = s; __syncthreads();
    for (int o = 128; o > 0; o >>= 1) { if (t < o) red[t] += red[t + o]; __syncthreads(); }
    if (t == 0) scores[row] = red[0] + b0[0];
}

// exact jax.lax.top_k semantics: top KCAP=4 of 8, ties broken by lower index
__global__ void topk_k(const float* __restrict__ scores, int* __restrict__ activeb,
                       int* __restrict__ exitm, float* __restrict__ capacc)
{
    int t = threadIdx.x;
    __shared__ int cnt[16];
    if (t < 16) {
        float s[8];
        #pragma unroll
        for (int l = 0; l < 8; ++l) s[l] = scores[t * 8 + l];
        int c = 0;
        #pragma unroll
        for (int l = 0; l < 8; ++l) {
            int rank = 0;
            #pragma unroll
            for (int m = 0; m < 8; ++m)
                if (s[m] > s[l] || (s[m] == s[l] && m < l)) rank++;
            int cont = (rank < 4) ? 1 : 0;
            int idx = t * 8 + l;
            exitm[idx]   = (activeb[idx] && !cont) ? 1 : 0;
            activeb[idx] = cont;
            c += cont;
        }
        cnt[t] = c;
    }
    __syncthreads();
    if (t == 0) {
        int tot = 0;
        for (int i = 0; i < 16; ++i) tot += cnt[i];
        float d = (float)tot / 128.f - 0.5f;
        capacc[0] += d * d;
    }
}

__global__ void exit_accum(const float* __restrict__ x, const int* __restrict__ exitm,
                           float* __restrict__ acc)
{
    int i = blockIdx.x * 256 + threadIdx.x;
    if (i < 131072) acc[i] += x[i] * (exitm[i >> 10] ? 1.f : 0.f);
}

__global__ __launch_bounds__(256)
void layernorm_k(const float* __restrict__ x, const float* __restrict__ g,
                 const float* __restrict__ bta, float* __restrict__ y)
{
    int row = blockIdx.x, t = threadIdx.x;
    __shared__ float r1[256], r2[256];
    const float* xr = x + (size_t)row * 1024;
    float s = 0.f, s2 = 0.f;
    for (int d = t; d < 1024; d += 256) { float v = xr[d]; s += v; s2 += v * v; }
    r1[t] = s; r2[t] = s2; __syncthreads();
    for (int o = 128; o > 0; o >>= 1) {
        if (t < o) { r1[t] += r1[t + o]; r2[t] += r2[t + o]; }
        __syncthreads();
    }
    float mean = r1[0] * (1.f / 1024.f);
    float var  = r2[0] * (1.f / 1024.f) - mean * mean;
    float inv  = rsqrtf(var + 1e-5f);
    for (int d = t; d < 1024; d += 256) {
        float v = xr[d];
        y[(size_t)row * 1024 + d] = (v - mean) * inv * g[d] + bta[d];
    }
}

// cross-attention: one block per (b, h, l); 8 latents x 2048 keys, hd=64
__global__ __launch_bounds__(256)
void ca_attn(const float* __restrict__ q, const float* __restrict__ Kb,
             const float* __restrict__ Vb, float* __restrict__ o)
{
    int blk = blockIdx.x;
    int l = blk & 7, h = (blk >> 3) & 15, b = blk >> 7;
    int t = threadIdx.x;
    __shared__ float qs[64];
    __shared__ float sc[2048];
    __shared__ float red[256];
    __shared__ float part[4][64];

    if (t < 64) qs[t] = q[(size_t)(b * 8 + l) * 1024 + h * 64 + t];
    __syncthreads();

    for (int s = t; s < 2048; s += 256) {
        const float* kr = Kb + ((size_t)b * 2048 + s) * 1024 + h * 64;
        float dot = 0.f;
        #pragma unroll 16
        for (int d = 0; d < 64; ++d) dot += qs[d] * kr[d];
        sc[s] = dot * 0.125f;
    }
    __syncthreads();

    float m = -3.4e38f;
    for (int s = t; s < 2048; s += 256) m = fmaxf(m, sc[s]);
    red[t] = m; __syncthreads();
    for (int o2 = 128; o2 > 0; o2 >>= 1) { if (t < o2) red[t] = fmaxf(red[t], red[t + o2]); __syncthreads(); }
    m = red[0]; __syncthreads();

    float sm = 0.f;
    for (int s = t; s < 2048; s += 256) { float e = expf(sc[s] - m); sc[s] = e; sm += e; }
    red[t] = sm; __syncthreads();
    for (int o2 = 128; o2 > 0; o2 >>= 1) { if (t < o2) red[t] += red[t + o2]; __syncthreads(); }
    float inv = 1.f / red[0];

    int d = t & 63, c = t >> 6;
    float a = 0.f;
    const float* vb = Vb + ((size_t)b * 2048 + c * 512) * 1024 + h * 64 + d;
    for (int s = 0; s < 512; ++s) a += sc[c * 512 + s] * vb[(size_t)s * 1024];
    part[c][d] = a;
    __syncthreads();
    if (t < 64)
        o[(size_t)(b * 8 + l) * 1024 + h * 64 + t] =
            (part[0][t] + part[1][t] + part[2][t] + part[3][t]) * inv;
}

// self-attention on 8 latents; one block per (b, h)
__global__ __launch_bounds__(64)
void sa_attn(const float* __restrict__ qkv, float* __restrict__ o)
{
    int b = blockIdx.x >> 4, h = blockIdx.x & 15;
    int t = threadIdx.x;
    __shared__ float qs[8][64], ks[8][64], vs[8][64], at[8][8];
    for (int idx = t; idx < 512; idx += 64) {
        int l = idx >> 6, d = idx & 63;
        size_t rb = (size_t)(b * 8 + l) * 3072;
        qs[l][d] = qkv[rb + h * 64 + d];
        ks[l][d] = qkv[rb + 1024 + h * 64 + d];
        vs[l][d] = qkv[rb + 2048 + h * 64 + d];
    }
    __syncthreads();
    {
        int l = t >> 3, s = t & 7;
        float dot = 0.f;
        #pragma unroll 16
        for (int d = 0; d < 64; ++d) dot += qs[l][d] * ks[s][d];
        at[l][s] = dot * 0.125f;
    }
    __syncthreads();
    if (t < 8) {
        float m = at[t][0];
        for (int s = 1; s < 8; ++s) m = fmaxf(m, at[t][s]);
        float sm = 0.f;
        for (int s = 0; s < 8; ++s) { float e = expf(at[t][s] - m); at[t][s] = e; sm += e; }
        float inv = 1.f / sm;
        for (int s = 0; s < 8; ++s) at[t][s] *= inv;
    }
    __syncthreads();
    for (int idx = t; idx < 512; idx += 64) {
        int l = idx >> 6, d = idx & 63;
        float s = 0.f;
        #pragma unroll
        for (int k = 0; k < 8; ++k) s += at[l][k] * vs[k][d];
        o[(size_t)(b * 8 + l) * 1024 + h * 64 + d] = s;
    }
}

__global__ void gelu_k(float* __restrict__ hbuf, int n)
{
    int i = blockIdx.x * 256 + threadIdx.x;
    if (i < n) { float v = hbuf[i]; hbuf[i] = 0.5f * v * (1.f + erff(v * 0.70710678118654752f)); }
}

__global__ __launch_bounds__(256)
void finalize_rows(const float* __restrict__ x, const float* __restrict__ acc,
                   const int* __restrict__ activeb, const float* __restrict__ osc,
                   float* __restrict__ out, float* __restrict__ rowsum,
                   float* __restrict__ rowsumsq)
{
    int row = blockIdx.x, t = threadIdx.x;
    __shared__ float red[256];
    const float* ar = acc + (size_t)row * 1024;
    const float* xr = x + (size_t)row * 1024;
    float am = activeb[row] ? 1.f : 0.f;
    float av[4], s2 = 0.f;
    #pragma unroll
    for (int j = 0; j < 4; ++j) { int d = t + 256 * j; float v = ar[d] + xr[d] * am; av[j] = v; s2 += v * v; }
    red[t] = s2; __syncthreads();
    for (int o = 128; o > 0; o >>= 1) { if (t < o) red[t] += red[t + o]; __syncthreads(); }
    float rms = sqrtf(red[0] * (1.f / 1024.f) + 1e-8f);
    float sc = osc[0] / rms;
    float ls = 0.f, ls2 = 0.f;
    #pragma unroll
    for (int j = 0; j < 4; ++j) {
        int d = t + 256 * j;
        float v = av[j] * sc;
        out[(size_t)row * 1024 + d] = v;
        ls += v; ls2 += v * v;
    }
    __syncthreads(); red[t] = ls; __syncthreads();
    for (int o = 128; o > 0; o >>= 1) { if (t < o) red[t] += red[t + o]; __syncthreads(); }
    if (t == 0) rowsum[row] = red[0];
    __syncthreads(); red[t] = ls2; __syncthreads();
    for (int o = 128; o > 0; o >>= 1) { if (t < o) red[t] += red[t + o]; __syncthreads(); }
    if (t == 0) rowsumsq[row] = red[0];
}

__global__ void finalize_scalars(const float* __restrict__ capacc, const float* __restrict__ rowsum,
                                 const float* __restrict__ rowsumsq, float* __restrict__ out)
{
    if (threadIdx.x == 0 && blockIdx.x == 0) {
        float s = 0.f, s2 = 0.f;
        for (int r = 0; r < 128; ++r) { s += rowsum[r]; s2 += rowsumsq[r]; }
        const float n = 131072.f;
        float mean = s / n;
        out[131072] = capacc[0] * 0.01f;
        out[131073] = 1.0f;
        out[131074] = (s2 - n * mean * mean) / (n - 1.f);
    }
}

extern "C" void kernel_launch(void* const* d_in, const int* in_sizes, int n_in,
                              void* d_out, int out_size, void* d_ws, size_t ws_size,
                              hipStream_t stream)
{
    (void)in_sizes; (void)n_in; (void)out_size; (void)ws_size;
    const float* src      = (const float*)d_in[0];
    const float* in_w     = (const float*)d_in[1];
    const float* in_b     = (const float*)d_in[2];
    const float* latents  = (const float*)d_in[3];
    const float* ca_in_w  = (const float*)d_in[4];
    const float* ca_in_b  = (const float*)d_in[5];
    const float* ca_out_w = (const float*)d_in[6];
    const float* ca_out_b = (const float*)d_in[7];
    const float* sa_in_w  = (const float*)d_in[8];
    const float* sa_in_b  = (const float*)d_in[9];
    const float* sa_out_w = (const float*)d_in[10];
    const float* sa_out_b = (const float*)d_in[11];
    const float* ffn_w1   = (const float*)d_in[12];
    const float* ffn_b1   = (const float*)d_in[13];
    const float* ffn_w2   = (const float*)d_in[14];
    const float* ffn_b2   = (const float*)d_in[15];
    const float* ln1_g    = (const float*)d_in[16];
    const float* ln1_b    = (const float*)d_in[17];
    const float* ln2_g    = (const float*)d_in[18];
    const float* ln2_b    = (const float*)d_in[19];
    const float* ln3_g    = (const float*)d_in[20];
    const float* ln3_b    = (const float*)d_in[21];
    const float* router_w = (const float*)d_in[22];
    const float* router_b = (const float*)d_in[23];
    const float* outscale = (const float*)d_in[24];
    float* out = (float*)d_out;

    // workspace carve (256B aligned)
    char* p = (char*)d_ws;
    auto take = [&](size_t n) { char* r = p; p += (n + 255) & ~(size_t)255; return r; };
    const size_t MS = 32768;  // B*S
    float*  keys  = (float*)take(MS * 1024 * 4);
    float*  Kb    = (float*)take(MS * 1024 * 4);
    float*  Vb    = (float*)take(MS * 1024 * 4);
    bf16_t* wbuf  = (bf16_t*)take(4194304 * 2);
    float*  x     = (float*)take(131072 * 4);
    float*  xn    = (float*)take(131072 * 4);
    float*  acc   = (float*)take(131072 * 4);
    float*  qb    = (float*)take(131072 * 4);
    float*  ob    = (float*)take(131072 * 4);
    float*  qkv   = (float*)take(393216 * 4);
    float*  hb    = (float*)take(524288 * 4);
    float*  scoreb   = (float*)take(128 * 4);
    int*    exitm    = (int*)take(128 * 4);
    int*    activeb  = (int*)take(128 * 4);
    float*  capacc   = (float*)take(256);
    float*  rowsum   = (float*)take(128 * 4);
    float*  rowsumsq = (float*)take(128 * 4);

    dim3 blk(256);

    // keys = src @ in_w^T + in_b   (32768 x 1024, K=4096)
    cvt_bf16<<<16384, blk, 0, stream>>>(in_w, wbuf, 4194304);
    wmma_gemm<false><<<dim3(8, 256), blk, 0, stream>>>(src, wbuf, in_b, nullptr, keys, (int)MS, 1024, 4096);
    init_state<<<512, blk, 0, stream>>>(latents, x, acc, activeb, capacc);

    for (int i = 0; i < 4; ++i) {
        const size_t iD = (size_t)i * 1024, i3D = (size_t)i * 3072, i4D = (size_t)i * 4096;
        const size_t iDD = (size_t)i * 1048576, i3DD = (size_t)i * 3145728, i4DD = (size_t)i * 4194304;

        router_k<<<128, blk, 0, stream>>>(x, router_w + iD, router_b + i, scoreb);
        topk_k<<<1, 64, 0, stream>>>(scoreb, activeb, exitm, capacc);
        exit_accum<<<512, blk, 0, stream>>>(x, exitm, acc);

        // ---- cross attention ----
        layernorm_k<<<128, blk, 0, stream>>>(x, ln1_g + iD, ln1_b + iD, xn);
        cvt_bf16<<<12288, blk, 0, stream>>>(ca_in_w + i3DD, wbuf, 3145728);
        wmma_gemm<false><<<dim3(8, 1),   blk, 0, stream>>>(xn,   wbuf,           ca_in_b + i3D,        nullptr, qb, 128, 1024, 1024);
        wmma_gemm<false><<<dim3(8, 256), blk, 0, stream>>>(keys, wbuf + 1048576, ca_in_b + i3D + 1024, nullptr, Kb, (int)MS, 1024, 1024);
        wmma_gemm<false><<<dim3(8, 256), blk, 0, stream>>>(keys, wbuf + 2097152, ca_in_b + i3D + 2048, nullptr, Vb, (int)MS, 1024, 1024);
        ca_attn<<<2048, blk, 0, stream>>>(qb, Kb, Vb, ob);
        cvt_bf16<<<4096, blk, 0, stream>>>(ca_out_w + iDD, wbuf, 1048576);
        wmma_gemm<true><<<dim3(8, 1), blk, 0, stream>>>(ob, wbuf, ca_out_b + iD, x, x, 128, 1024, 1024);

        // ---- self attention ----
        layernorm_k<<<128, blk, 0, stream>>>(x, ln2_g + iD, ln2_b + iD, xn);
        cvt_bf16<<<12288, blk, 0, stream>>>(sa_in_w + i3DD, wbuf, 3145728);
        wmma_gemm<false><<<dim3(24, 1), blk, 0, stream>>>(xn, wbuf, sa_in_b + i3D, nullptr, qkv, 128, 3072, 1024);
        sa_attn<<<256, 64, 0, stream>>>(qkv, ob);
        cvt_bf16<<<4096, blk, 0, stream>>>(sa_out_w + iDD, wbuf, 1048576);
        wmma_gemm<true><<<dim3(8, 1), blk, 0, stream>>>(ob, wbuf, sa_out_b + iD, x, x, 128, 1024, 1024);

        // ---- FFN ----
        layernorm_k<<<128, blk, 0, stream>>>(x, ln3_g + iD, ln3_b + iD, xn);
        cvt_bf16<<<16384, blk, 0, stream>>>(ffn_w1 + i4DD, wbuf, 4194304);
        wmma_gemm<false><<<dim3(32, 1), blk, 0, stream>>>(xn, wbuf, ffn_b1 + i4D, nullptr, hb, 128, 4096, 1024);
        gelu_k<<<2048, blk, 0, stream>>>(hb, 524288);
        cvt_bf16<<<16384, blk, 0, stream>>>(ffn_w2 + i4DD, wbuf, 4194304);
        wmma_gemm<true><<<dim3(8, 1), blk, 0, stream>>>(hb, wbuf, ffn_b2 + iD, x, x, 128, 1024, 4096);
    }

    finalize_rows<<<128, blk, 0, stream>>>(x, acc, activeb, outscale, out, rowsum, rowsumsq);
    finalize_scalars<<<1, 1, 0, stream>>>(capacc, rowsum, rowsumsq, out);
}
